// GAT_7507602833557
// MI455X (gfx1250) — matrix operand order repaired and checked by reading, not compiled, not run on hardware
//
#include <hip/hip_runtime.h>
#include <hip/hip_bf16.h>

#define ALPHA_SLOPE 0.2f
#define NEG_FILL   -9.0e15f
#define LOG2E       1.4426950408889634f

constexpr int NN     = 4096;
constexpr int NIN    = 64;
constexpr int NHID   = 32;
constexpr int NHEADS = 8;
constexpr int HSP    = 6;

typedef __attribute__((ext_vector_type(16))) _Float16 v16h;
typedef __attribute__((ext_vector_type(8)))  float    v8f;
typedef __attribute__((ext_vector_type(4)))  float    v4f;

// Raw v_exp_f32 (natively 2^x on AMD) — one VALU transcendental, no wrapper.
__device__ __forceinline__ float fast_exp2(float x) {
    return __builtin_amdgcn_exp2f(x);
}

// ---------------------------------------------------------------------------
// K1: per-head projections Wh = x @ W  (stored transposed as f16 for WMMA B),
//     plus attention pre-scores f1/f2 (spatial: Wh·a, intent: embeds·a).
//     f1/f2 are prescaled by log2(e): scores live in the log2 domain so the
//     softmax uses bare v_exp_f32 (natively 2^x) with no extra multiply.
// ---------------------------------------------------------------------------
__global__ void proj_kernel(const float* __restrict__ x,
                            const float* __restrict__ ie,
                            const float* __restrict__ W_sp,
                            const float* __restrict__ a_sp,
                            const float* __restrict__ W_int,
                            const float* __restrict__ a_int,
                            _Float16* __restrict__ WhT,   // [8][32][4096] f16
                            float* __restrict__ f1,       // [8][4096]  (log2 dom)
                            float* __restrict__ f2)       // [8][4096]  (log2 dom)
{
    int idx = blockIdx.x * blockDim.x + threadIdx.x;   // 8*4096*32 threads
    int f = idx & 31;
    int n = (idx >> 5) & (NN - 1);
    int h = idx >> 17;
    const float* W = (h < HSP) ? (W_sp + h * NIN * NHID)
                               : (W_int + (h - HSP) * NIN * NHID);
    const float* xr = x + (size_t)n * NIN;
    float acc = 0.f;
#pragma unroll
    for (int k = 0; k < NIN; ++k) acc = fmaf(xr[k], W[k * NHID + f], acc);
    WhT[(size_t)(h * NHID + f) * NN + n] = (_Float16)acc;

    float v1, v2;
    if (h < HSP) {
        v1 = acc * a_sp[h * 2 * NHID + f];
        v2 = acc * a_sp[h * 2 * NHID + NHID + f];
    } else {
        float e = ie[(size_t)n * 32 + f];
        v1 = e * a_int[(h - HSP) * 64 + f];
        v2 = e * a_int[(h - HSP) * 64 + 32 + f];
    }
#pragma unroll
    for (int off = 16; off > 0; off >>= 1) {
        v1 += __shfl_xor(v1, off, 32);
        v2 += __shfl_xor(v2, off, 32);
    }
    if (f == 0) {
        f1[(size_t)h * NN + n] = v1 * LOG2E;
        f2[(size_t)h * NN + n] = v2 * LOG2E;
    }
}

// ---------------------------------------------------------------------------
// K2a: exact per-row softmax statistics (log2 domain).
//   All 32 lanes sweep ONE row together (lane*4 + step*128 columns): every
//   adj load is 512B fully-coalesced, L2-resident. Per-lane online (m,l) over
//   its column subset; one 5-step butterfly merge per row at the end.
//   No LDS, no per-tile cross-lane traffic.
// ---------------------------------------------------------------------------
__global__ void softmax_stats_kernel(const float* __restrict__ adj,
                                     const float* __restrict__ f1s, // [H][N]
                                     const float* __restrict__ f2s, // [H][N]
                                     float* __restrict__ mbuf,      // [H][N]
                                     float* __restrict__ lbuf)      // [H][N]
{
    const int lane = threadIdx.x & 31;
    const int wv   = threadIdx.x >> 5;
    const int head = blockIdx.y;
    const int r0   = blockIdx.x * 64 + wv * 8;    // 8 waves * 8 rows / block
    const float* f1h = f1s + (size_t)head * NN;
    const float* f2h = f2s + (size_t)head * NN;

    for (int r = 0; r < 8; ++r) {
        const int row = r0 + r;
        const float f1r = f1h[row];
        const float* arow = adj + (size_t)row * NN;
        float m = NEG_FILL, l = 0.f;
        for (int j = lane * 4; j < NN; j += 128) {
            v4f av = *(const v4f*)(arow + j);
            v4f fv = *(const v4f*)(f2h + j);
            float s[4];
#pragma unroll
            for (int k = 0; k < 4; ++k) {
                float t = f1r + fv[k];
                t = fmaxf(t, ALPHA_SLOPE * t);            // leaky_relu
                s[k] = (av[k] > 0.f) ? t : NEG_FILL;      // mask fill
            }
            float cm = fmaxf(fmaxf(s[0], s[1]), fmaxf(s[2], s[3]));
            float mn = fmaxf(m, cm);
            float ss = fast_exp2(s[0] - mn) + fast_exp2(s[1] - mn)
                     + fast_exp2(s[2] - mn) + fast_exp2(s[3] - mn);
            l = l * fast_exp2(m - mn) + ss;
            m = mn;
        }
        // merge 32 per-lane partial (m,l) pairs
#pragma unroll
        for (int off = 16; off > 0; off >>= 1) {
            float mo = __shfl_xor(m, off, 32);
            float lo = __shfl_xor(l, off, 32);
            float mn = fmaxf(m, mo);
            l = l * fast_exp2(m - mn) + lo * fast_exp2(mo - mn);
            m = mn;
        }
        if (lane == 0) {
            mbuf[(size_t)head * NN + row] = m;
            lbuf[(size_t)head * NN + row] = l;
        }
    }
}

// ---------------------------------------------------------------------------
// K2b: WMMA aggregation pass. With (m,l) precomputed, the normalized
//   probability p = 2^(s-m)/l is produced directly in the 16-bit A-fragment
//   layout used here (lane l<16: row l, K 0..15; lane l+16: row l, K
//   16..31) and fed straight to v_wmma_f32_16x16x32_f16. No LDS, no
//   cross-lane ops, no accumulator rescale -> no dscnt waits / hazard NOPs.
//   B operand = pre-transposed f16 values BT[head][f][n]: 32B contiguous/lane.
// ---------------------------------------------------------------------------
template<int F, int ACT>   // ACT: 0 = ELU (hidden heads), 1 = tanh (output)
__global__ void flash_gat_kernel(const float* __restrict__ adj,
                                 const float* __restrict__ f1s,  // [H][N]
                                 const float* __restrict__ f2s,  // [H][N]
                                 const float* __restrict__ mbuf, // [H][N]
                                 const float* __restrict__ lbuf, // [H][N]
                                 const _Float16* __restrict__ BT,// [H][F][N]
                                 float* __restrict__ out,        // [N][outStride]
                                 int outStride)
{
    constexpr int NFRAG = F / 16;
    const int lane = threadIdx.x & 31;
    const int wv   = threadIdx.x >> 5;
    const int head = blockIdx.y;
    const int i0   = blockIdx.x * 128 + wv * 16;   // 8 waves * 16 rows
    const int row  = lane & 15;
    const int hi   = lane >> 4;                    // half-wave selector

    const float* f2h = f2s + (size_t)head * NN;
    const _Float16* BTh = BT + (size_t)head * F * NN;

    const float f1r  = f1s[(size_t)head * NN + i0 + row];
    const float mrow = mbuf[(size_t)head * NN + i0 + row];
    const float linv = 1.0f / lbuf[(size_t)head * NN + i0 + row];
    const float* arow = adj + (size_t)(i0 + row) * NN;

    v8f acc[NFRAG];
#pragma unroll
    for (int q = 0; q < NFRAG; ++q)
#pragma unroll
        for (int v = 0; v < 8; ++v) acc[q][v] = 0.f;

    for (int j0 = 0; j0 < NN; j0 += 32) {
        const int cb = j0 + hi * 16;
        // adjacency + f2 for this lane's 16 fragment columns (one 16-elt run)
        v4f a0 = *(const v4f*)(arow + cb);
        v4f a1 = *(const v4f*)(arow + cb + 4);
        v4f a2 = *(const v4f*)(arow + cb + 8);
        v4f a3 = *(const v4f*)(arow + cb + 12);
        v4f q0 = *(const v4f*)(f2h + cb);
        v4f q1 = *(const v4f*)(f2h + cb + 4);
        v4f q2 = *(const v4f*)(f2h + cb + 8);
        v4f q3 = *(const v4f*)(f2h + cb + 12);

        v16h afrag;
#pragma unroll
        for (int k = 0; k < 16; ++k) {
            float av = (k < 4) ? a0[k] : (k < 8) ? a1[k - 4]
                     : (k < 12) ? a2[k - 8] : a3[k - 12];
            float fv = (k < 4) ? q0[k] : (k < 8) ? q1[k - 4]
                     : (k < 12) ? q2[k - 8] : q3[k - 12];
            float s = f1r + fv;
            s = fmaxf(s, ALPHA_SLOPE * s);                // leaky_relu
            s = (av > 0.f) ? s : NEG_FILL;                // mask fill
            afrag[k] = (_Float16)(fast_exp2(s - mrow) * linv); // normalized prob
        }
        // B fragments: lane holds column (lane&15)+16q, 16 consecutive K halves
#pragma unroll
        for (int q = 0; q < NFRAG; ++q) {
            const _Float16* bp = BTh + (size_t)(q * 16 + (lane & 15)) * NN
                               + j0 + hi * 16;
            v16h bfrag = *(const v16h*)bp;
            acc[q] = __builtin_amdgcn_wmma_f32_16x16x32_f16(
                false, afrag, false, bfrag, (short)0, acc[q], false, false);
        }
    }

    // finalize: already normalized; activation + scatter (C row = v + 8*hi)
#pragma unroll
    for (int v = 0; v < 8; ++v) {
        int orow = i0 + v + 8 * hi;
#pragma unroll
        for (int q = 0; q < NFRAG; ++q) {
            float val = acc[q][v];
            if (ACT == 0) val = (val > 0.f) ? val : (__expf(val) - 1.0f); // elu
            else          val = tanhf(val);
            out[(size_t)orow * outStride + head * F + q * 16 + (lane & 15)] = val;
        }
    }
}

// ---------------------------------------------------------------------------
// K3: Who = h @ W_out  ([4096,256]x[256,64]) + f16 transpose for WMMA B.
// ---------------------------------------------------------------------------
__global__ void outproj_kernel(const float* __restrict__ hbuf,
                               const float* __restrict__ W_out,
                               float* __restrict__ Who,
                               _Float16* __restrict__ WhoT)
{
    int idx = blockIdx.x * blockDim.x + threadIdx.x;   // 4096*64 threads
    int f = idx & 63;
    int n = idx >> 6;
    const float* hr = hbuf + (size_t)n * 256;
    float acc = 0.f;
#pragma unroll 8
    for (int k = 0; k < 256; ++k) acc = fmaf(hr[k], W_out[k * 64 + f], acc);
    Who[(size_t)n * 64 + f]  = acc;
    WhoT[(size_t)f * NN + n] = (_Float16)acc;
}

// ---------------------------------------------------------------------------
// K4: o1/o2 = Who · a_out halves (log2-prescaled), one wave32 per node.
// ---------------------------------------------------------------------------
__global__ void oscore_kernel(const float* __restrict__ Who,
                              const float* __restrict__ a_out,
                              float* __restrict__ o1,
                              float* __restrict__ o2)
{
    int lane = threadIdx.x & 31;
    int n = blockIdx.x * (blockDim.x >> 5) + (threadIdx.x >> 5);
    float w0 = Who[(size_t)n * 64 + lane];
    float w1 = Who[(size_t)n * 64 + 32 + lane];
    float v1 = w0 * a_out[lane]      + w1 * a_out[32 + lane];
    float v2 = w0 * a_out[64 + lane] + w1 * a_out[96 + lane];
#pragma unroll
    for (int off = 16; off > 0; off >>= 1) {
        v1 += __shfl_xor(v1, off, 32);
        v2 += __shfl_xor(v2, off, 32);
    }
    if (lane == 0) { o1[n] = v1 * LOG2E; o2[n] = v2 * LOG2E; }
}

// ---------------------------------------------------------------------------
extern "C" void kernel_launch(void* const* d_in, const int* in_sizes, int n_in,
                              void* d_out, int out_size, void* d_ws, size_t ws_size,
                              hipStream_t stream) {
    (void)in_sizes; (void)n_in; (void)out_size; (void)ws_size;
    const float* x     = (const float*)d_in[0];
    const float* adj   = (const float*)d_in[1];
    const float* ie    = (const float*)d_in[2];
    const float* W_sp  = (const float*)d_in[3];
    const float* a_sp  = (const float*)d_in[4];
    const float* W_int = (const float*)d_in[5];
    const float* a_int = (const float*)d_in[6];
    const float* W_out = (const float*)d_in[7];
    const float* a_out = (const float*)d_in[8];
    float* out = (float*)d_out;

    char* ws = (char*)d_ws;
    size_t off = 0;
    auto alloc = [&](size_t bytes) -> void* {
        void* p = ws + off;
        off = (off + bytes + 255) & ~(size_t)255;
        return p;
    };
    _Float16* WhT  = (_Float16*)alloc((size_t)NHEADS * NHID * NN * 2); // 2 MB
    float*    f1s  = (float*)   alloc((size_t)NHEADS * NN * 4);
    float*    f2s  = (float*)   alloc((size_t)NHEADS * NN * 4);
    float*    mb1  = (float*)   alloc((size_t)NHEADS * NN * 4);
    float*    lb1  = (float*)   alloc((size_t)NHEADS * NN * 4);
    float*    hbuf = (float*)   alloc((size_t)NN * 256 * 4);           // 4 MB
    float*    Who  = (float*)   alloc((size_t)NN * 64 * 4);
    _Float16* WhoT = (_Float16*)alloc((size_t)64 * NN * 2);
    float*    o1   = (float*)   alloc((size_t)NN * 4);
    float*    o2   = (float*)   alloc((size_t)NN * 4);
    float*    mb2  = (float*)   alloc((size_t)NN * 4);
    float*    lb2  = (float*)   alloc((size_t)NN * 4);

    // K1: projections + log2-domain f1/f2
    proj_kernel<<<dim3((NHEADS * NN * 32) / 256), 256, 0, stream>>>(
        x, ie, W_sp, a_sp, W_int, a_int, WhT, f1s, f2s);

    // K2a: exact row softmax stats for 8 heads
    softmax_stats_kernel<<<dim3(NN / 64, NHEADS), 256, 0, stream>>>(
        adj, f1s, f2s, mb1, lb1);

    // K2b: 8-head WMMA aggregation, F=32, ELU -> concat h[4096,256]
    flash_gat_kernel<32, 0><<<dim3(NN / 128, NHEADS), 256, 0, stream>>>(
        adj, f1s, f2s, mb1, lb1, WhT, hbuf, 256);

    // K3: Who = h @ W_out  (+ f16 transpose)
    outproj_kernel<<<dim3((NN * 64) / 256), 256, 0, stream>>>(
        hbuf, W_out, Who, WhoT);

    // K4: o1/o2 scores (log2-prescaled)
    oscore_kernel<<<dim3(NN / 8), 256, 0, stream>>>(Who, a_out, o1, o2);

    // K5a: output-layer softmax stats
    softmax_stats_kernel<<<dim3(NN / 64, 1), 256, 0, stream>>>(
        adj, o1, o2, mb2, lb2);

    // K5b: output WMMA aggregation, F=64, tanh -> d_out
    flash_gat_kernel<64, 1><<<dim3(NN / 128, 1), 256, 0, stream>>>(
        adj, o1, o2, mb2, lb2, WhoT, out, 64);
}
